// BandPassFilterRaw_3143916061221
// MI455X (gfx1250) — matrix operand order repaired and checked by reading, not compiled, not run on hardware
//
#include <hip/hip_runtime.h>
#include <math.h>

typedef __attribute__((ext_vector_type(16))) __bf16 v16bf;
typedef __attribute__((ext_vector_type(8)))  float  v8f;

union Frag16 { unsigned int u[8]; v16bf v; };

__device__ inline unsigned short f2bf(float f) {
  unsigned int u = __float_as_uint(f);
  u += 0x7FFFu + ((u >> 16) & 1u);           // round-to-nearest-even
  return (unsigned short)(u >> 16);
}

__device__ inline v8f wmma_bf(const Frag16& a, const Frag16& b, v8f c) {
  return __builtin_amdgcn_wmma_f32_16x16x32_bf16(false, a.v, false, b.v,
                                                 (short)0, c, false, false);
}

constexpr int NB = 32, HH = 128, WW = 128, WF = 65, BC = 64;
constexpr int KP = 80;    // padded k-extent for masked spectrum / A (5 N-tiles)
constexpr int KP2 = 96;   // padded K for row-transform GEMM (3 K-chunks of 32)
constexpr float TWO_PI = 6.28318530717958647692f;

// ---------------------------------------------------------------- tables ----
__global__ void gen_tables(float* cosT, float* sinT,
                           unsigned short* ChbT, unsigned short* ShbT,
                           unsigned short* CfbT, unsigned short* SfbT) {
  int g0 = blockIdx.x * blockDim.x + threadIdx.x;
  int stride = gridDim.x * blockDim.x;
  for (int idx = g0; idx < HH * HH; idx += stride) {
    int a = idx >> 7, b = idx & 127;
    float ph = (TWO_PI / 128.f) * (float)((a * b) & 127);
    float c = cosf(ph), s = sinf(ph);
    cosT[idx] = c; sinT[idx] = s;
    ChbT[idx] = f2bf(c); ShbT[idx] = f2bf(s);
  }
  // Row-transform tables, [w][k] layout (k pair-contiguous), Hermitian fold +
  // full ortho norm 1/128 baked in.  band = Ar@Cf + Ai@Sf.
  for (int idx = g0; idx < WW * KP2; idx += stride) {
    int w = idx / KP2, k = idx - w * KP2;
    float cv = 0.f, sv = 0.f;
    if (k < WF) {
      float f = (k == 0 || k == 64) ? 1.0f : 2.0f;
      float ph = (TWO_PI / 128.f) * (float)((k * w) & 127);
      cv =  f * cosf(ph) * (1.0f / 128.0f);
      sv = -f * sinf(ph) * (1.0f / 128.0f);
    }
    CfbT[idx] = f2bf(cv); SfbT[idx] = f2bf(sv);
  }
}

// --------------------------------------------- time embedding -> tbias[b,n] -
__global__ void time_bias(const int* times, const float* month_tab,
                          const float* day_tab, const float* hour_tab,
                          const float* pproj_w, const float* pproj_b,
                          const float* fc_t_w, const float* fc_t_b,
                          const float* fuse_w, const float* fuse_b,
                          float* tbias) {
  __shared__ float temb[8][72];
  __shared__ float tt[8][32];
  int tid = threadIdx.x;
  for (int idx = tid; idx < 8 * 72; idx += blockDim.x) {
    int b = idx / 72, d = idx - b * 72;
    int t = times[b];
    int mo = (t / 10000) % 100, dy = (t / 100) % 100, hr = t % 100;
    float e = month_tab[mo * 72 + d] + day_tab[dy * 72 + d] + hour_tab[hr * 72 + d];
    float acc = pproj_b[d];
    for (int j = 0; j < 72; ++j) {
      int f = j / 24, inn = j - f * 24;
      float per = (f == 0) ? 12.f : ((f == 1) ? 31.f : 24.f);
      float v = (f == 0) ? (float)mo : ((f == 1) ? (float)dy : (float)hr);
      float i_ = (float)((inn % 12) + 1);
      float ph = TWO_PI * (v / per) * i_;
      float pf = (inn < 12) ? sinf(ph) : cosf(ph);
      acc = fmaf(pf, pproj_w[d * 72 + j], acc);
    }
    temb[b][d] = e + acc;
  }
  __syncthreads();
  for (int idx = tid; idx < 8 * 32; idx += blockDim.x) {
    int b = idx / 32, n = idx - b * 32;
    float a = fc_t_b[n];
    for (int d = 0; d < 72; ++d) a = fmaf(fc_t_w[n * 72 + d], temb[b][d], a);
    tt[b][n] = fmaxf(a, 0.01f * a);
  }
  __syncthreads();
  for (int idx = tid; idx < 8 * 32; idx += blockDim.x) {
    int b = idx / 32, n = idx - b * 32;
    float s = fuse_b[n];
    for (int j = 0; j < 32; ++j) s = fmaf(fuse_w[n * 64 + 32 + j], tt[b][j], s);
    tbias[idx] = s;
  }
}

// -------------------------------------------------- forward rfft2 (fp32) ----
// Step A: row DFT  R1[bc,y,k] = sum_x x[bc,y,x] e^{-2pi i k x/W}
__global__ void fwdA(const float* x, const float* cosT, const float* sinT,
                     float* R1r, float* R1i) {
  int gid = blockIdx.x * blockDim.x + threadIdx.x;        // BC*HH*WF
  if (gid >= BC * HH * WF) return;
  int k = gid % WF; int rest = gid / WF;                   // rest = bc*HH + y
  const float* row = x + (size_t)rest * WW;
  const float* cr = cosT + k * HH;
  const float* sr = sinT + k * HH;
  float ar = 0.f, ai = 0.f;
  for (int xx = 0; xx < WW; ++xx) {
    float v = row[xx];
    ar = fmaf(v,  cr[xx], ar);
    ai = fmaf(v, -sr[xx], ai);
  }
  R1r[gid] = ar; R1i[gid] = ai;
}

// Step B: column DFT  G[bc,h',k] = (1/128) sum_y R1[y,k] e^{-2pi i h' y/H}
__global__ void fwdB(const float* R1r, const float* R1i,
                     const float* cosT, const float* sinT,
                     float* Gr, float* Gi) {
  int gid = blockIdx.x * blockDim.x + threadIdx.x;        // BC*HH*WF
  if (gid >= BC * HH * WF) return;
  int k = gid % WF; int rest = gid / WF;
  int hp = rest % HH; int bc = rest / HH;
  const float* rr = R1r + (size_t)bc * (HH * WF);
  const float* ri = R1i + (size_t)bc * (HH * WF);
  const float* cr = cosT + hp * HH;
  const float* sr = sinT + hp * HH;
  float gr = 0.f, gi = 0.f;
  for (int y = 0; y < HH; ++y) {
    float a = rr[y * WF + k], b = ri[y * WF + k];
    float c = cr[y], s = sr[y];
    gr = fmaf(a, c, fmaf(b,  s, gr));
    gi = fmaf(b, c, fmaf(a, -s, gi));
  }
  Gr[gid] = gr * (1.f / 128.f);
  Gi[gid] = gi * (1.f / 128.f);
}

// ---------------- fused per-band inverse DFT + softmax-weighted reduction ---
__global__ void __launch_bounds__(512)
band_kernel(const float* x, const int* band_id,
            const float* Gr, const float* Gi,
            const unsigned short* ChbT, const unsigned short* ShbT,
            const unsigned short* CfbT, const unsigned short* SfbT,
            const float* tbias, const float* fuse_w,
            const float* cnn_w, const float* cnn_b, float* y) {
  __shared__ unsigned short mGr[KP][HH];    // masked spectrum, [k][h'] (20 KB)
  __shared__ unsigned short mGi[KP][HH];    // (20 KB)
  __shared__ unsigned short Ar[HH][KP2];    // column-IDFT result, [h][k] (24 KB)
  __shared__ unsigned short Ai[HH][KP2];    // (24 KB)
  __shared__ unsigned short CfbS[WW][KP2];  // row-transform tables (24 KB)
  __shared__ unsigned short SfbS[WW][KP2];  // (24 KB)  -> 136 KB total LDS

  const int bc = blockIdx.x;
  const int b = bc >> 3;
  const int tid = threadIdx.x;
  const int lane = tid & 31;
  const int wv = tid >> 5;                 // wave 0..15
  const int lm = lane & 15, lh = lane >> 4;
  const float* xp  = x  + (size_t)bc * (HH * WW);
  const float* grp = Gr + (size_t)bc * (HH * WF);
  const float* gip = Gi + (size_t)bc * (HH * WF);

  // Stage the GEMM2 B-operand tables into LDS with async global->LDS copies
  // (ASYNCcnt-tracked; 16 B per lane per op, 3 ops per table per thread).
  {
    unsigned int cOff = (unsigned int)(size_t)&CfbS[0][0];
    unsigned int sOff = (unsigned int)(size_t)&SfbS[0][0];
    for (int q = 0; q < 3; ++q) {
      unsigned int byteIdx = (unsigned int)(tid + q * 512) * 16u;
      unsigned int ldc = cOff + byteIdx;
      unsigned int lds = sOff + byteIdx;
      const void* gc = (const char*)CfbT + byteIdx;
      const void* gs = (const char*)SfbT + byteIdx;
      asm volatile("global_load_async_to_lds_b128 %0, %1, off"
                   :: "v"(ldc), "v"(gc) : "memory");
      asm volatile("global_load_async_to_lds_b128 %0, %1, off"
                   :: "v"(lds), "v"(gs) : "memory");
    }
  }

  if (tid == 0) { __builtin_prefetch(ChbT, 0, 3); __builtin_prefetch(grp, 0, 3); }

  // zero LDS once: pad rows/cols never written again stay zero
  for (int i = tid; i < KP * HH; i += 512) { (&mGr[0][0])[i] = 0; (&mGi[0][0])[i] = 0; }
  for (int i = tid; i < HH * KP2; i += 512) { (&Ar[0][0])[i] = 0; (&Ai[0][0])[i] = 0; }

  // per-wave fixed output tiles (4 tiles of 16x16) + register accumulators
  float xr[4][8], num[4][8], den[4][8];
  for (int i = 0; i < 4; ++i) {
    int tt = wv * 4 + i, th = tt >> 3, tw = tt & 7;
    int wcol = tw * 16 + lm;
    for (int v = 0; v < 8; ++v) {
      int hrow = th * 16 + lh * 8 + v;
      xr[i][v] = xp[hrow * WW + wcol];
      num[i][v] = 0.f; den[i][v] = 0.f;
    }
  }
  asm volatile("s_wait_asynccnt 0x0" ::: "memory");   // own async copies done
  __syncthreads();                                    // all waves' copies visible

  for (int n = 0; n < NB; ++n) {
    // (a) masked spectrum -> LDS bf16, single predicated pass over all bins
    for (int t = tid; t < HH * WF; t += 512) {
      int hp = t / WF, k = t - hp * WF;
      bool on = (band_id[t] == n);
      mGr[k][hp] = f2bf(on ? grp[t] : 0.f);
      mGi[k][hp] = f2bf(on ? gip[t] : 0.f);
    }
    __syncthreads();

    // (b) GEMM1: A[h,k] = F_H (+i) @ mG ; 8 h-tiles x 5 k-tiles, K=128
    for (int t = wv; t < 40; t += 16) {
      int th = t / 5, tk = t - th * 5;
      v8f aCr = {}, aSi = {}, aCi = {}, aSr = {};
      for (int ch = 0; ch < 4; ++ch) {
        Frag16 fC, fS, fR, fI;
        int mBase = th * 16, kBase = ch * 32, nBase = tk * 16;
        for (int j = 0; j < 8; ++j) {
          int p = j & 3, g = j >> 2;
          int K0a = g * 16 + lh * 8 + p * 2;         // A-operand K packing
          int K0b = g * 16 + lh * 8 + p * 2;         // B-operand K packing
          int ia = (mBase + lm) * HH + kBase + K0a;
          fC.u[j] = *(const unsigned int*)(ChbT + ia);
          fS.u[j] = *(const unsigned int*)(ShbT + ia);
          fR.u[j] = *(const unsigned int*)(&mGr[nBase + lm][kBase + K0b]);
          fI.u[j] = *(const unsigned int*)(&mGi[nBase + lm][kBase + K0b]);
        }
        aCr = wmma_bf(fC, fR, aCr); aSi = wmma_bf(fS, fI, aSi);
        aCi = wmma_bf(fC, fI, aCi); aSr = wmma_bf(fS, fR, aSr);
      }
      for (int v = 0; v < 8; ++v) {                  // Ar = C@Gr - S@Gi, Ai = C@Gi + S@Gr
        int h = th * 16 + lh * 8 + v, k = tk * 16 + lm;
        Ar[h][k] = f2bf(aCr[v] - aSi[v]);
        Ai[h][k] = f2bf(aCi[v] + aSr[v]);
      }
    }
    __syncthreads();

    // (c) GEMM2 (row transform) + per-pixel score/exp + accumulate
    for (int i = 0; i < 4; ++i) {
      int tt = wv * 4 + i, th = tt >> 3, tw = tt & 7;
      v8f acc = {};
      for (int ch = 0; ch < 3; ++ch) {
        Frag16 fAr, fAi, fCc, fSs;
        int mBase = th * 16, kBase = ch * 32, nBase = tw * 16;
        for (int j = 0; j < 8; ++j) {
          int p = j & 3, g = j >> 2;
          int K0a = g * 16 + lh * 8 + p * 2;
          int K0b = g * 16 + lh * 8 + p * 2;
          fAr.u[j] = *(const unsigned int*)(&Ar[mBase + lm][kBase + K0a]);
          fAi.u[j] = *(const unsigned int*)(&Ai[mBase + lm][kBase + K0a]);
          fCc.u[j] = *(const unsigned int*)(&CfbS[nBase + lm][kBase + K0b]);
          fSs.u[j] = *(const unsigned int*)(&SfbS[nBase + lm][kBase + K0b]);
        }
        acc = wmma_bf(fAr, fCc, acc);
        acc = wmma_bf(fAi, fSs, acc);
      }
      float tb = tbias[b * NB + n];
      for (int v = 0; v < 8; ++v) {
        float xv = xr[i][v];
        float s = tb;
        for (int m = 0; m < NB; ++m) {
          float l = fmaf(xv, cnn_w[m], cnn_b[m]);
          l = fmaxf(l, 0.01f * l);                   // leaky-ReLU: mul+max
          s = fmaf(fuse_w[n * 64 + m], l, s);
        }
        float e = __expf(s);
        num[i][v] = fmaf(e, acc[v], num[i][v]);
        den[i][v] += e;
      }
    }
    __syncthreads();
  }

  for (int i = 0; i < 4; ++i) {
    int tt = wv * 4 + i, th = tt >> 3, tw = tt & 7;
    int wcol = tw * 16 + lm;
    for (int v = 0; v < 8; ++v) {
      int hrow = th * 16 + lh * 8 + v;
      y[(size_t)bc * (HH * WW) + hrow * WW + wcol] = 32.0f * num[i][v] / den[i][v];
    }
  }
}

// ------------------------------------------- channel LayerNorm + MLP epilogue
__global__ void ffn_kernel(const float* y, const float* fc1_w, const float* fc1_b,
                           const float* fc2_w, const float* fc2_b,
                           const float* ln_g, const float* ln_b, float* out) {
  int gid = blockIdx.x * blockDim.x + threadIdx.x;  // B*H*W pixels
  if (gid >= 8 * HH * WW) return;
  int hw = gid % (HH * WW); int b = gid / (HH * WW);
  const size_t plane = HH * WW;
  float yp[8];
  for (int c = 0; c < 8; ++c) yp[c] = y[((size_t)(b * 8 + c)) * plane + hw];
  float mu = 0.f; for (int c = 0; c < 8; ++c) mu += yp[c]; mu *= 0.125f;
  float var = 0.f;
  for (int c = 0; c < 8; ++c) { float d = yp[c] - mu; var = fmaf(d, d, var); }
  var *= 0.125f;
  float rstd = rsqrtf(var + 1e-5f);
  float ln[8];
  for (int c = 0; c < 8; ++c) ln[c] = (yp[c] - mu) * rstd * ln_g[c] + ln_b[c];
  float h1[8];
  for (int o = 0; o < 8; ++o) {
    float a = fc1_b[o];
    for (int c = 0; c < 8; ++c) a = fmaf(fc1_w[o * 8 + c], ln[c], a);
    h1[o] = 0.5f * a * (1.0f + erff(a * 0.70710678118654752440f));
  }
  for (int o = 0; o < 8; ++o) {
    float a = fc2_b[o];
    for (int c = 0; c < 8; ++c) a = fmaf(fc2_w[o * 8 + c], h1[c], a);
    out[((size_t)(b * 8 + o)) * plane + hw] = yp[o] + a;
  }
}

// ---------------------------------------------------------------- launcher --
extern "C" void kernel_launch(void* const* d_in, const int* in_sizes, int n_in,
                              void* d_out, int out_size, void* d_ws, size_t ws_size,
                              hipStream_t stream) {
  const float* x        = (const float*)d_in[0];
  const int*   times    = (const int*)  d_in[1];
  const float* cnn_w    = (const float*)d_in[2];
  const float* cnn_b    = (const float*)d_in[3];
  const float* fc_t_w   = (const float*)d_in[4];
  const float* fc_t_b   = (const float*)d_in[5];
  const float* fuse_w   = (const float*)d_in[6];
  const float* fuse_b   = (const float*)d_in[7];
  const float* fc1_w    = (const float*)d_in[8];
  const float* fc1_b    = (const float*)d_in[9];
  const float* fc2_w    = (const float*)d_in[10];
  const float* fc2_b    = (const float*)d_in[11];
  const float* ln_g     = (const float*)d_in[12];
  const float* ln_b     = (const float*)d_in[13];
  const float* month_tab= (const float*)d_in[14];
  const float* day_tab  = (const float*)d_in[15];
  const float* hour_tab = (const float*)d_in[16];
  const float* pproj_w  = (const float*)d_in[17];
  const float* pproj_b  = (const float*)d_in[18];
  const int*   band_id  = (const int*)  d_in[19];
  float* out = (float*)d_out;

  char* ws = (char*)d_ws;
  size_t o = 0;
  float*          cosT = (float*)(ws + o);          o += (size_t)HH * HH * 4;       // 64 KB
  float*          sinT = (float*)(ws + o);          o += (size_t)HH * HH * 4;
  unsigned short* ChbT = (unsigned short*)(ws + o); o += (size_t)HH * HH * 2;       // 32 KB
  unsigned short* ShbT = (unsigned short*)(ws + o); o += (size_t)HH * HH * 2;
  unsigned short* CfbT = (unsigned short*)(ws + o); o += (size_t)WW * KP2 * 2;      // 24 KB
  unsigned short* SfbT = (unsigned short*)(ws + o); o += (size_t)WW * KP2 * 2;
  float*          tb   = (float*)(ws + o);          o += (size_t)8 * NB * 4;
  o = (o + 255) & ~(size_t)255;
  float* GrW  = (float*)(ws + o); o += (size_t)BC * HH * WF * 4;                    // 2.1 MB
  float* GiW  = (float*)(ws + o); o += (size_t)BC * HH * WF * 4;
  float* R1r  = (float*)(ws + o); o += (size_t)BC * HH * WF * 4;
  float* R1i  = (float*)(ws + o); o += (size_t)BC * HH * WF * 4;
  float* yW   = (float*)(ws + o); o += (size_t)BC * HH * WW * 4;                    // 4 MB

  gen_tables<<<64, 256, 0, stream>>>(cosT, sinT, ChbT, ShbT, CfbT, SfbT);
  time_bias<<<1, 576, 0, stream>>>(times, month_tab, day_tab, hour_tab,
                                   pproj_w, pproj_b, fc_t_w, fc_t_b,
                                   fuse_w, fuse_b, tb);
  fwdA<<<(BC * HH * WF + 255) / 256, 256, 0, stream>>>(x, cosT, sinT, R1r, R1i);
  fwdB<<<(BC * HH * WF + 255) / 256, 256, 0, stream>>>(R1r, R1i, cosT, sinT, GrW, GiW);
  band_kernel<<<BC, 512, 0, stream>>>(x, band_id, GrW, GiW, ChbT, ShbT, CfbT, SfbT,
                                      tb, fuse_w, cnn_w, cnn_b, yW);
  ffn_kernel<<<(8 * HH * WW + 255) / 256, 256, 0, stream>>>(yW, fc1_w, fc1_b,
                                                            fc2_w, fc2_b,
                                                            ln_g, ln_b, out);
}
